// AttentionBetweenWordsAndChars_5772436046213
// MI455X (gfx1250) — matrix operand order repaired and checked by reading, not compiled, not run on hardware
//
#include <hip/hip_runtime.h>

// B=64, W=512, D_IN=300, H=512. Output [B, W, 2*H] f32.
#define SEQW     512
#define DIN      300
#define HDIM     512
#define KT_COUNT 10      // K padded 300 -> 320 = 10 * 32
#define NT_COUNT 32      // 512 / 16
#define ASTRIDE  328     // LDS A row stride (16B-aligned, bank-spread: 36*l mod 64 distinct)

typedef __bf16 v16bf __attribute__((ext_vector_type(16)));
typedef __bf16 v8bf  __attribute__((ext_vector_type(8)));
typedef float  v8f   __attribute__((ext_vector_type(8)));

__device__ __forceinline__ unsigned short f2bf(float f) {
  unsigned int u = __float_as_uint(f);
  u += 0x7FFFu + ((u >> 16) & 1u);          // round-to-nearest-even
  return (unsigned short)(u >> 16);
}
__device__ __forceinline__ float bf2f(unsigned short h) {
  return __uint_as_float(((unsigned int)h) << 16);
}
__device__ __forceinline__ v16bf cat16(v8bf lo, v8bf hi) {
  return __builtin_shufflevector(lo, hi, 0,1,2,3,4,5,6,7,8,9,10,11,12,13,14,15);
}

// ---------------------------------------------------------------------------
// Kernel 0: pack Wk,Wq,Wv (f32 row-major [300][512]) into the exact per-lane
// bf16 WMMA B-fragment image: wp[((mat*10+kt)*32+nt)*512 + lane*16 + e],
// element e of lane L holds W[kt*32 + 16*(L>=16) + e][nt*16 + (L&15)].
// K in [300,320) zero-filled. Main kernel then loads each B frag with two
// coalesced global_load_b128 (32B/lane, 1KB/wave contiguous).
// ---------------------------------------------------------------------------
__global__ __launch_bounds__(256) void pack_weights_bf16(
    const float* __restrict__ Wk, const float* __restrict__ Wq,
    const float* __restrict__ Wv, unsigned short* __restrict__ wp) {
  unsigned int f = blockIdx.x * 256u + threadIdx.x;
  if (f >= 3u * KT_COUNT * NT_COUNT * 32u * 16u) return;
  unsigned int e    = f & 15u;
  unsigned int lane = (f >> 4) & 31u;
  unsigned int nt   = (f >> 9) & 31u;
  unsigned int kt   = (f >> 14) % KT_COUNT;
  unsigned int mat  = (f >> 14) / KT_COUNT;
  unsigned int k = kt * 32u + ((lane >> 4) << 4) + e;
  unsigned int n = nt * 16u + (lane & 15u);
  const float* Wm = (mat == 0) ? Wk : ((mat == 1) ? Wq : Wv);
  unsigned short v = 0;
  if (k < DIN) v = f2bf(Wm[(size_t)k * HDIM + n]);
  wp[f] = v;
}

// ---------------------------------------------------------------------------
// Kernel 1: fused per-16-word tile.
//   A tile [32 x 300]: rows 0..15 = word_vectors, 16..31 = char_vectors.
//   Phase 1: K = A@Wk+bk, Q = A@Wq+bq via v_wmma_f32_16x16x32_bf16 -> LDS bf16
//   Phase 2: per-word 2x2 logits (dot over H), softmax
//   Phase 3: Amix[qi-row] = a0*A_word + a1*A_char  (fold attn@V into input mix)
//   Phase 4: Out = Amix@Wv + bv  -> global f32 directly
// ---------------------------------------------------------------------------
__global__ __launch_bounds__(256) void fused_word_char_attn(
    const float* __restrict__ charv, const float* __restrict__ wordv,
    const float* __restrict__ bk, const float* __restrict__ bq,
    const float* __restrict__ bv, const unsigned short* __restrict__ wp,
    float* __restrict__ out) {
  __shared__ __align__(16) unsigned short As [32][ASTRIDE];
  __shared__ __align__(16) unsigned short Ams[32][ASTRIDE];
  __shared__ unsigned short Ks[32][HDIM];
  __shared__ unsigned short Qs[32][HDIM];
  __shared__ float sPart[64][4];
  __shared__ float attnS[16][2][2];

  const int  tid    = threadIdx.x;
  const int  lane   = tid & 31;
  const int  wave   = tid >> 5;              // 8 waves
  const bool hiHalf = (lane >= 16);
  const int  nc     = lane & 15;             // column within N tile
  const int  m0     = lane & 15;             // A-frag row within M tile
  const long g0     = (long)blockIdx.x * 16; // first flat word index

  // ---- Phase 0: stage A (f32 -> bf16) into LDS, zero-pad K to 328 ----------
  {
    int r  = tid >> 3;                       // 32 rows, 8 threads per row
    int c0 = tid & 7;
    const float* src = (r < 16) ? (wordv + (size_t)(g0 + r) * DIN)
                                : (charv + (size_t)(g0 + (r - 16)) * DIN);
    const float4* s4 = (const float4*)src;   // 300 % 4 == 0, 16B-aligned rows
    for (int q = c0; q < DIN / 4; q += 8) {
      float4 x = s4[q];
      As[r][4*q+0] = f2bf(x.x);  As[r][4*q+1] = f2bf(x.y);
      As[r][4*q+2] = f2bf(x.z);  As[r][4*q+3] = f2bf(x.w);
    }
    for (int c = DIN + c0; c < ASTRIDE; c += 8) As[r][c] = 0;
  }
  __syncthreads();

  // ---- Phase 1: K and Q projections -> LDS bf16 ----------------------------
  for (int mat = 0; mat < 2; ++mat) {
    const float* bias = (mat == 0) ? bk : bq;
    unsigned short (*dst)[HDIM] = (mat == 0) ? Ks : Qs;
    for (int j = 0; j < 4; ++j) {
      const int nt = wave * 4 + j;
      v8f acc0 = {};                          // rows 0..15  (word)
      v8f acc1 = {};                          // rows 16..31 (char)
      #pragma unroll
      for (int kt = 0; kt < KT_COUNT; ++kt) {
        const int kb = kt * 32 + (hiHalf ? 16 : 0);
        v16bf a0 = cat16(*(const v8bf*)&As[m0][kb],
                         *(const v8bf*)&As[m0][kb + 8]);
        v16bf a1 = cat16(*(const v8bf*)&As[16 + m0][kb],
                         *(const v8bf*)&As[16 + m0][kb + 8]);
        const unsigned short* bp =
            wp + ((((mat * KT_COUNT) + kt) * NT_COUNT + nt) << 9) + (lane << 4);
        v16bf bfrag = cat16(*(const v8bf*)bp, *(const v8bf*)(bp + 8));
        acc0 = __builtin_amdgcn_wmma_f32_16x16x32_bf16(false, a0, false, bfrag,
                                                       (short)0, acc0, false, false);
        acc1 = __builtin_amdgcn_wmma_f32_16x16x32_bf16(false, a1, false, bfrag,
                                                       (short)0, acc1, false, false);
      }
      const int   n = nt * 16 + nc;
      const float b = bias[n];
      #pragma unroll
      for (int v = 0; v < 8; ++v) {
        const int r = v + (hiHalf ? 8 : 0);   // C/D layout: M = v + 8*(lane>=16)
        dst[r][n]      = f2bf(acc0[v] + b);
        dst[16 + r][n] = f2bf(acc1[v] + b);
      }
    }
  }
  __syncthreads();

  // ---- Phase 2: per-word 2x2 logits + softmax ------------------------------
  {
    const int t    = tid >> 2;                // 64 tasks = word(16) x qi(2) x ki(2)
    const int part = tid & 3;
    const int i  = t >> 2, qi = (t >> 1) & 1, ki = t & 1;
    const int qr = qi * 16 + i, kr = ki * 16 + i;
    float s = 0.f;
    const int n0 = part * 128;
    for (int n = n0; n < n0 + 128; ++n)
      s += bf2f(Qs[qr][n]) * bf2f(Ks[kr][n]);
    sPart[t][part] = s;
  }
  __syncthreads();
  if (tid < 16) {
    const float inv_temp = 0.04419417382415922f;   // 1/sqrt(512)
    float sm[2][2];
    #pragma unroll
    for (int qi = 0; qi < 2; ++qi)
      #pragma unroll
      for (int ki = 0; ki < 2; ++ki) {
        const int t = (tid << 2) | (qi << 1) | ki;
        sm[qi][ki] = (sPart[t][0] + sPart[t][1] + sPart[t][2] + sPart[t][3]) * inv_temp;
      }
    #pragma unroll
    for (int qi = 0; qi < 2; ++qi) {
      const float m  = fmaxf(sm[qi][0], sm[qi][1]);
      const float e0 = __expf(sm[qi][0] - m);
      const float e1 = __expf(sm[qi][1] - m);
      const float iv = 1.f / (e0 + e1);
      attnS[tid][qi][0] = e0 * iv;
      attnS[tid][qi][1] = e1 * iv;
    }
  }
  __syncthreads();

  // ---- Phase 3: attention-mixed A rows (folds attn @ V) --------------------
  for (int u = tid; u < 32 * ASTRIDE; u += 256) {
    const int r = u / ASTRIDE, c = u - r * ASTRIDE;
    const int i = r & 15, qi = r >> 4;
    const float aw = attnS[i][qi][0], ac = attnS[i][qi][1];
    Ams[r][c] = f2bf(aw * bf2f(As[i][c]) + ac * bf2f(As[16 + i][c]));
  }
  __syncthreads();

  // ---- Phase 4: Out = Amix @ Wv + bv -> global -----------------------------
  for (int j = 0; j < 4; ++j) {
    const int nt = wave * 4 + j;
    v8f acc0 = {};                            // rows 0..15  -> qi = 0 outputs
    v8f acc1 = {};                            // rows 16..31 -> qi = 1 outputs
    #pragma unroll
    for (int kt = 0; kt < KT_COUNT; ++kt) {
      const int kb = kt * 32 + (hiHalf ? 16 : 0);
      v16bf a0 = cat16(*(const v8bf*)&Ams[m0][kb],
                       *(const v8bf*)&Ams[m0][kb + 8]);
      v16bf a1 = cat16(*(const v8bf*)&Ams[16 + m0][kb],
                       *(const v8bf*)&Ams[16 + m0][kb + 8]);
      const unsigned short* bp =
          wp + (((2 * KT_COUNT + kt) * NT_COUNT + nt) << 9) + (lane << 4);
      v16bf bfrag = cat16(*(const v8bf*)bp, *(const v8bf*)(bp + 8));
      acc0 = __builtin_amdgcn_wmma_f32_16x16x32_bf16(false, a0, false, bfrag,
                                                     (short)0, acc0, false, false);
      acc1 = __builtin_amdgcn_wmma_f32_16x16x32_bf16(false, a1, false, bfrag,
                                                     (short)0, acc1, false, false);
    }
    const int   n = nt * 16 + nc;
    const float b = bv[n];
    #pragma unroll
    for (int v = 0; v < 8; ++v) {
      const int r = v + (hiHalf ? 8 : 0);     // word index within tile
      out[(size_t)(g0 + r) * 1024 + n]       = acc0[v] + b;  // qi = 0 half
      out[(size_t)(g0 + r) * 1024 + 512 + n] = acc1[v] + b;  // qi = 1 half
    }
  }
}

// ---------------------------------------------------------------------------
extern "C" void kernel_launch(void* const* d_in, const int* in_sizes, int n_in,
                              void* d_out, int out_size, void* d_ws, size_t ws_size,
                              hipStream_t stream) {
  const float* charv = (const float*)d_in[0];
  const float* wordv = (const float*)d_in[1];
  const float* Wk    = (const float*)d_in[2];
  const float* bk    = (const float*)d_in[3];
  const float* Wq    = (const float*)d_in[4];
  const float* bq    = (const float*)d_in[5];
  const float* Wv    = (const float*)d_in[6];
  const float* bv    = (const float*)d_in[7];
  unsigned short* wpack = (unsigned short*)d_ws;  // needs 983,040 B (3*10*32*512*2)

  // 3 * 10 * 32 * 32 * 16 = 491,520 packed bf16 elements
  pack_weights_bf16<<<1920, 256, 0, stream>>>(Wk, Wq, Wv, wpack);
  // 32768 words / 16 words-per-block = 2048 blocks
  fused_word_char_attn<<<2048, 256, 0, stream>>>(charv, wordv, bk, bq, bv, wpack,
                                                 (float*)d_out);
}
